// Qwen3Attention_22127671509142
// MI455X (gfx1250) — matrix operand order repaired and checked by reading, not compiled, not run on hardware
//
#include <hip/hip_runtime.h>

#define B_   2
#define T_   2048
#define H_   2048
#define NH_  16
#define NKV_ 8
#define HD_  128

typedef __attribute__((ext_vector_type(8)))  __bf16 v8bf;
typedef __attribute__((ext_vector_type(16))) __bf16 v16bf;
typedef __attribute__((ext_vector_type(8)))  float  v8f;
typedef __attribute__((ext_vector_type(4)))  float  v4f;
typedef __attribute__((ext_vector_type(4)))  unsigned int u32x4;
typedef __attribute__((ext_vector_type(8)))  int i32x8;
typedef __attribute__((ext_vector_type(4)))  int i32x4;

#if __has_builtin(__builtin_amdgcn_tensor_load_to_lds) && \
    __has_builtin(__builtin_amdgcn_s_wait_tensorcnt)
#define USE_TDM 1
#endif

static __device__ __forceinline__ v16bf cat16(v8bf lo, v8bf hi) {
  return __builtin_shufflevector(lo, hi, 0,1,2,3,4,5,6,7,8,9,10,11,12,13,14,15);
}
static __device__ __forceinline__ v8f vzero8() {
  v8f z = {0.f,0.f,0.f,0.f,0.f,0.f,0.f,0.f};
  return z;
}
#define WMMA_BF16(a,b,c) \
  __builtin_amdgcn_wmma_f32_16x16x32_bf16(false,(a),false,(b),(short)0,(c),false,false)

// ---------------------------------------------------------------------------
// f32 -> bf16 conversion, 4 elements per thread
// ---------------------------------------------------------------------------
__global__ void cvt_f32_to_bf16_x4(const float* __restrict__ src,
                                   __bf16* __restrict__ dst, int n4) {
  int i = blockIdx.x * blockDim.x + threadIdx.x;
  if (i < n4) {
    v4f v = *(const v4f*)(src + (size_t)i * 4);
    __bf16* d = dst + (size_t)i * 4;
    d[0] = (__bf16)v.x; d[1] = (__bf16)v.y;
    d[2] = (__bf16)v.z; d[3] = (__bf16)v.w;
  }
}

// ---------------------------------------------------------------------------
// src [K][N] f32 (row-major)  ->  dst [N][K] bf16 (row-major, i.e. transposed)
// ---------------------------------------------------------------------------
__global__ void transpose_cvt_kernel(const float* __restrict__ src,
                                     __bf16* __restrict__ dst, int K, int N) {
  int n = blockIdx.x * 16 + threadIdx.x;
  int k = blockIdx.y * 16 + threadIdx.y;
  if (n < N && k < K)
    dst[(size_t)n * K + k] = (__bf16)src[(size_t)k * N + n];
}

// ---------------------------------------------------------------------------
// C[M,N] (f32) = A[M,K] (bf16, row-major) * Bt[N,K]^T (bf16, row-major)
// Block: 256 thr = 8 waves (4 along M x 2 along N); wave tile 32x64;
// block tile 128x128. WMMA 16x16x32 bf16, K-step 32.
// ---------------------------------------------------------------------------
__global__ __launch_bounds__(256) void gemm_bf16_wmma(
    const __bf16* __restrict__ A, const __bf16* __restrict__ Bt,
    float* __restrict__ C, int M, int N, int K) {
  const int tid  = threadIdx.x;
  const int wave = tid >> 5, lane = tid & 31;
  const int l16 = lane & 15, lh = lane >> 4;
  const int wm = wave & 3, wn = wave >> 2;
  const int m0 = blockIdx.y * 128 + wm * 32;
  const int n0 = blockIdx.x * 128 + wn * 64;

  v8f acc[2][4];
  #pragma unroll
  for (int i = 0; i < 2; i++)
    #pragma unroll
    for (int j = 0; j < 4; j++) acc[i][j] = vzero8();

  const __bf16* aRow0 = A + (size_t)(m0 + l16) * K + lh * 16;
  const __bf16* aRow1 = A + (size_t)(m0 + 16 + l16) * K + lh * 16;
  const __bf16* bRow[4];
  #pragma unroll
  for (int j = 0; j < 4; j++)
    bRow[j] = Bt + (size_t)(n0 + j * 16 + l16) * K + lh * 16;

  for (int k0 = 0; k0 < K; k0 += 32) {
    v16bf af0 = *(const v16bf*)(aRow0 + k0);
    v16bf af1 = *(const v16bf*)(aRow1 + k0);
    __builtin_prefetch(aRow0 + k0 + 64, 0, 1);
    __builtin_prefetch(bRow[0] + k0 + 64, 0, 1);
    #pragma unroll
    for (int j = 0; j < 4; j++) {
      v16bf bf = *(const v16bf*)(bRow[j] + k0);
      acc[0][j] = WMMA_BF16(af0, bf, acc[0][j]);
      acc[1][j] = WMMA_BF16(af1, bf, acc[1][j]);
    }
  }

  #pragma unroll
  for (int i = 0; i < 2; i++)
    #pragma unroll
    for (int j = 0; j < 4; j++) {
      const int n = n0 + j * 16 + l16;
      #pragma unroll
      for (int v = 0; v < 8; v++) {
        const int m = m0 + i * 16 + v + 8 * lh;
        C[(size_t)m * N + n] = acc[i][j][v];
      }
    }
}

// ---------------------------------------------------------------------------
// Per-(b,t,head) RMSNorm + RoPE + relayout to head-major bf16.
// Grid (8, B*T), block 128 = 4 waves; one wave per head-job (32 jobs/(b,t)).
// Lane l owns dims {2l, 2l+1, 2l+64, 2l+65} so RoPE pairs stay in-lane.
// ---------------------------------------------------------------------------
__global__ __launch_bounds__(128) void normrope_kernel(
    const float* __restrict__ QKV, const float* __restrict__ qw,
    const float* __restrict__ kw, __bf16* __restrict__ Qh,
    __bf16* __restrict__ Kh, __bf16* __restrict__ Vh) {
  const int bt = blockIdx.y;
  const int b = bt / T_, t = bt % T_;
  const int wave = threadIdx.x >> 5, lane = threadIdx.x & 31;
  const int job = blockIdx.x * 4 + wave;
  const int i0 = 2 * lane, i1 = 2 * lane + 1;

  const float* src; __bf16* dst; int mode;
  if (job < NH_) {
    mode = 0;
    src = QKV + (size_t)bt * 4096 + job * HD_;
    dst = Qh + (((size_t)b * NH_ + job) * T_ + t) * HD_;
  } else if (job < NH_ + NKV_) {
    mode = 1; const int kh = job - NH_;
    src = QKV + (size_t)bt * 4096 + 2048 + kh * HD_;
    dst = Kh + (((size_t)b * NKV_ + kh) * T_ + t) * HD_;
  } else {
    mode = 2; const int kh = job - NH_ - NKV_;
    src = QKV + (size_t)bt * 4096 + 3072 + kh * HD_;
    dst = Vh + (((size_t)b * NKV_ + kh) * T_ + t) * HD_;
  }

  float x0 = src[i0], x1 = src[i1], y0 = src[i0 + 64], y1 = src[i1 + 64];
  if (mode != 2) {
    float ss = x0 * x0 + x1 * x1 + y0 * y0 + y1 * y1;
    ss += __shfl_xor(ss, 16, 32);
    ss += __shfl_xor(ss, 8, 32);
    ss += __shfl_xor(ss, 4, 32);
    ss += __shfl_xor(ss, 2, 32);
    ss += __shfl_xor(ss, 1, 32);
    const float r = rsqrtf(ss * (1.0f / 128.0f) + 1e-6f);
    const float* w = (mode == 0) ? qw : kw;
    x0 *= r * w[i0]; x1 *= r * w[i1];
    y0 *= r * w[i0 + 64]; y1 *= r * w[i1 + 64];
    const float kfre = 13.815510557964274f / 64.0f;  // ln(1e6)/64
    float s0, c0, s1, c1;
    __sincosf((float)t * __expf(-(float)i0 * kfre), &s0, &c0);
    __sincosf((float)t * __expf(-(float)i1 * kfre), &s1, &c1);
    const float o0 = x0 * c0 - y0 * s0, o1 = x1 * c1 - y1 * s1;
    const float o2 = y0 * c0 + x0 * s0, o3 = y1 * c1 + x1 * s1;
    x0 = o0; x1 = o1; y0 = o2; y1 = o3;
  }
  dst[i0] = (__bf16)x0; dst[i1] = (__bf16)x1;
  dst[i0 + 64] = (__bf16)y0; dst[i1 + 64] = (__bf16)y1;
}

// ---------------------------------------------------------------------------
// Causal GQA flash attention. Grid (T/64, B*NH). Block 128 = 4 waves; each
// wave owns 16 q-rows. K tile staged via the Tensor Data Mover (one
// 128x64-elem 2-D tile DMA, TENSORcnt-tracked); V tile is register-
// transposed into LDS. Q@K^T and P@V use v_wmma_f32_16x16x32_bf16;
// online softmax via 16-lane shuffles.
// ---------------------------------------------------------------------------
__global__ __launch_bounds__(128) void flash_attn_kernel(
    const __bf16* __restrict__ Qh, const __bf16* __restrict__ Kh,
    const __bf16* __restrict__ Vh, __bf16* __restrict__ ctx) {
  const int qt = blockIdx.x;
  const int bh = blockIdx.y;
  const int b = bh >> 4, h = bh & 15;
  const int kvh = h >> 1;                // n_rep = NH/NKV = 2
  const int wave = threadIdx.x >> 5, lane = threadIdx.x & 31;
  const int l16 = lane & 15, lh = lane >> 4;
  const int q0 = qt * 64;
  const int qrow0 = q0 + wave * 16;

  __shared__ __bf16 Ktile[64 * 128] __attribute__((aligned(32)));
  __shared__ __bf16 Vt[128 * 64]    __attribute__((aligned(32)));
  __shared__ __bf16 Ptile[4][16 * 64] __attribute__((aligned(32)));

  const __bf16* Qbase = Qh + ((size_t)b * NH_ + h) * T_ * HD_;
  const __bf16* Kbase = Kh + ((size_t)b * NKV_ + kvh) * T_ * HD_;
  const __bf16* Vbase = Vh + ((size_t)b * NKV_ + kvh) * T_ * HD_;

  v16bf qf[4];
  #pragma unroll
  for (int kc = 0; kc < 4; kc++) {
    const __bf16* qp = Qbase + (size_t)(qrow0 + l16) * HD_ + kc * 32 + lh * 16;
    qf[kc] = *(const v16bf*)qp;
  }

  v8f o[8];
  #pragma unroll
  for (int d = 0; d < 8; d++) o[d] = vzero8();
  float mrow[8], lrow[8];
  #pragma unroll
  for (int v = 0; v < 8; v++) { mrow[v] = -1e30f; lrow[v] = 0.f; }

  const float scale = 0.08838834764831845f;  // 1/sqrt(128)
  const int nkt = qt + 1;
  for (int kt = 0; kt < nkt; ++kt) {
    __syncthreads();
    {
#if defined(USE_TDM)
      // --- K tile via Tensor Data Mover: 2-D tile, 128 elems x 64 rows,
      //     2 bytes/elem, row stride 128 elems, into LDS Ktile. ---
      if (wave == 0) {
        const unsigned ldsoff = (unsigned)(unsigned long long)(&Ktile[0]);
        const unsigned long long ga =
            (unsigned long long)(Kbase + (size_t)(kt * 64) * HD_);
        u32x4 g0;
        g0.x = 1u;                                    // count=1 (valid, user)
        g0.y = ldsoff;                                // lds_addr
        g0.z = (unsigned)(ga & 0xffffffffu);          // global_addr[31:0]
        g0.w = (unsigned)((ga >> 32) & 0x01ffffffu)   // global_addr[56:32]
               | (2u << 30);                          // type=2 ("image")
        i32x8 g1;
        g1[0] = 0x00010000;          // data_size=1 -> 2 bytes/element
        g1[1] = (int)(128u << 16);   // tensor_dim0[15:0]=128 @ bits[31:16]
        g1[2] = (int)(64u << 16);    // tensor_dim0 hi=0 | tensor_dim1 lo=64
        g1[3] = (int)(128u << 16);   // tensor_dim1 hi=0 | tile_dim0=128
        g1[4] = 64;                  // tile_dim1=64 | tile_dim2=0
        g1[5] = 128;                 // tensor_dim0_stride[31:0]=128
        g1[6] = 0;                   // stride hi | dim1_stride lo
        g1[7] = 0;                   // dim1_stride hi
        i32x4 gz4 = {0, 0, 0, 0};    // groups 2/3 unused (<=2-D tensor)
        i32x8 gz8 = {0, 0, 0, 0, 0, 0, 0, 0};
        __builtin_amdgcn_tensor_load_to_lds(g0, g1, gz4, gz4, gz8, 0);
        __builtin_amdgcn_s_wait_tensorcnt(0);
      }
#else
      // --- K tile: manual global -> LDS copy ---
      {
        const int r  = threadIdx.x >> 1;
        const int c0 = (threadIdx.x & 1) * 64;
        const __bf16* ksrc = Kbase + (size_t)(kt * 64 + r) * HD_ + c0;
        __bf16* kdst = Ktile + r * 128 + c0;
        #pragma unroll
        for (int u = 0; u < 8; u++)
          *(v8bf*)(kdst + u * 8) = *(const v8bf*)(ksrc + u * 8);
      }
#endif
      // --- V tile: register transpose into Vt[d][key] (TDM can't transpose)
      const int r  = threadIdx.x >> 1;
      const int c0 = (threadIdx.x & 1) * 64;
      const __bf16* vsrc = Vbase + (size_t)(kt * 64 + r) * HD_ + c0;
      #pragma unroll
      for (int u8 = 0; u8 < 8; ++u8) {
        v8bf vv = *(const v8bf*)(vsrc + u8 * 8);
        #pragma unroll
        for (int e = 0; e < 8; ++e)
          Vt[(c0 + u8 * 8 + e) * 64 + r] = vv[e];
      }
    }
    __syncthreads();

    // S = Q @ K^T (16 q-rows x 64 keys per wave)
    v8f s[4];
    #pragma unroll
    for (int nt = 0; nt < 4; nt++) s[nt] = vzero8();
    #pragma unroll
    for (int kc = 0; kc < 4; kc++) {
      #pragma unroll
      for (int nt = 0; nt < 4; nt++) {
        v16bf kb = *(const v16bf*)(Ktile + (nt * 16 + l16) * 128 + kc * 32 + lh * 16);
        s[nt] = WMMA_BF16(qf[kc], kb, s[nt]);
      }
    }

    // scale + causal mask (only diagonal tile needs it)
    float pv[4][8];
    const bool diag = (kt == qt);
    #pragma unroll
    for (int nt = 0; nt < 4; nt++) {
      const int key = kt * 64 + nt * 16 + l16;
      #pragma unroll
      for (int v = 0; v < 8; v++) {
        float xv = s[nt][v] * scale;
        if (diag && key > (qrow0 + v + 8 * lh)) xv = -1e30f;
        pv[nt][v] = xv;
      }
    }

    // online softmax: row max across 4 tiles + 16 lanes
    float mt[8];
    #pragma unroll
    for (int v = 0; v < 8; v++)
      mt[v] = fmaxf(fmaxf(pv[0][v], pv[1][v]), fmaxf(pv[2][v], pv[3][v]));
    #pragma unroll
    for (int v = 0; v < 8; v++) {
      mt[v] = fmaxf(mt[v], __shfl_xor(mt[v], 1, 32));
      mt[v] = fmaxf(mt[v], __shfl_xor(mt[v], 2, 32));
      mt[v] = fmaxf(mt[v], __shfl_xor(mt[v], 4, 32));
      mt[v] = fmaxf(mt[v], __shfl_xor(mt[v], 8, 32));
    }
    float cf[8];
    #pragma unroll
    for (int v = 0; v < 8; v++) {
      const float mn = fmaxf(mrow[v], mt[v]);
      cf[v] = __expf(mrow[v] - mn);
      mrow[v] = mn;
    }
    #pragma unroll
    for (int d = 0; d < 8; d++)
      #pragma unroll
      for (int v = 0; v < 8; v++) o[d][v] *= cf[v];

    float rs[8];
    #pragma unroll
    for (int v = 0; v < 8; v++) rs[v] = 0.f;
    #pragma unroll
    for (int nt = 0; nt < 4; nt++)
      #pragma unroll
      for (int v = 0; v < 8; v++) {
        const float e = __expf(pv[nt][v] - mrow[v]);
        pv[nt][v] = e;
        rs[v] += e;
      }
    #pragma unroll
    for (int v = 0; v < 8; v++) {
      rs[v] += __shfl_xor(rs[v], 1, 32);
      rs[v] += __shfl_xor(rs[v], 2, 32);
      rs[v] += __shfl_xor(rs[v], 4, 32);
      rs[v] += __shfl_xor(rs[v], 8, 32);
      lrow[v] = lrow[v] * cf[v] + rs[v];
    }

    // P -> wave-private LDS (C-layout -> A-fragment layout change)
    #pragma unroll
    for (int nt = 0; nt < 4; nt++)
      #pragma unroll
      for (int v = 0; v < 8; v++)
        Ptile[wave][(v + 8 * lh) * 64 + nt * 16 + l16] = (__bf16)pv[nt][v];

    // O += P @ V   (keys = K-dim, 2 steps of 32; 8 d-tiles of 16)
    #pragma unroll
    for (int kc2 = 0; kc2 < 2; kc2++) {
      const __bf16* pp = &Ptile[wave][l16 * 64 + kc2 * 32 + lh * 16];
      v16bf pf = *(const v16bf*)pp;
      #pragma unroll
      for (int dt = 0; dt < 8; dt++) {
        v16bf vf = *(const v16bf*)(Vt + (dt * 16 + l16) * 64 + kc2 * 32 + lh * 16);
        o[dt] = WMMA_BF16(pf, vf, o[dt]);
      }
    }
  }

  #pragma unroll
  for (int v = 0; v < 8; v++) lrow[v] = 1.0f / lrow[v];
  #pragma unroll
  for (int dt = 0; dt < 8; dt++) {
    const int d = dt * 16 + l16;
    #pragma unroll
    for (int v = 0; v < 8; v++) {
      const int qr = qrow0 + v + 8 * lh;
      ctx[((size_t)b * T_ + qr) * (NH_ * HD_) + h * HD_ + d] =
          (__bf16)(o[dt][v] * lrow[v]);
    }
  }
}

// ---------------------------------------------------------------------------
extern "C" void kernel_launch(void* const* d_in, const int* in_sizes, int n_in,
                              void* d_out, int out_size, void* d_ws, size_t ws_size,
                              hipStream_t stream) {
  (void)in_sizes; (void)n_in; (void)out_size; (void)ws_size;
  const float* x  = (const float*)d_in[0];
  // d_in[1] = attn_mask (causal; applied analytically)
  const float* Wq = (const float*)d_in[2];
  const float* Wk = (const float*)d_in[3];
  const float* Wv = (const float*)d_in[4];
  const float* Wo = (const float*)d_in[5];
  const float* qw = (const float*)d_in[6];
  const float* kw = (const float*)d_in[7];
  float* out = (float*)d_out;

  char* ws = (char*)d_ws;
  size_t off = 0;
  auto wsAlloc = [&](size_t bytes) {
    char* p = ws + off;
    off += (bytes + 255) & ~(size_t)255;
    return p;
  };
  __bf16* x_bf   = (__bf16*)wsAlloc((size_t)B_ * T_ * H_ * 2);        // 16 MB
  __bf16* Wqkv_t = (__bf16*)wsAlloc((size_t)4096 * 2048 * 2);          // 16 MB
  __bf16* Wo_t   = (__bf16*)wsAlloc((size_t)2048 * 2048 * 2);          //  8 MB
  float*  QKV    = (float*) wsAlloc((size_t)4096 * 4096 * 4);          // 64 MB
  __bf16* Qh     = (__bf16*)wsAlloc((size_t)B_ * NH_  * T_ * HD_ * 2); // 16 MB
  __bf16* Kh     = (__bf16*)wsAlloc((size_t)B_ * NKV_ * T_ * HD_ * 2); //  8 MB
  __bf16* Vh     = (__bf16*)wsAlloc((size_t)B_ * NKV_ * T_ * HD_ * 2); //  8 MB
  __bf16* ctxb   = (__bf16*)wsAlloc((size_t)4096 * 2048 * 2);          // 16 MB

  // 1) x -> bf16
  {
    const int n4 = B_ * T_ * H_ / 4;
    cvt_f32_to_bf16_x4<<<(n4 + 255) / 256, 256, 0, stream>>>(x, x_bf, n4);
  }
  // 2) weights: transpose + convert (fused QKV weight: rows 0..2047 = Wq^T,
  //    2048..3071 = Wk^T, 3072..4095 = Wv^T)
  {
    dim3 blk(16, 16);
    transpose_cvt_kernel<<<dim3(2048 / 16, 2048 / 16), blk, 0, stream>>>(
        Wq, Wqkv_t, 2048, 2048);
    transpose_cvt_kernel<<<dim3(1024 / 16, 2048 / 16), blk, 0, stream>>>(
        Wk, Wqkv_t + (size_t)2048 * 2048, 2048, 1024);
    transpose_cvt_kernel<<<dim3(1024 / 16, 2048 / 16), blk, 0, stream>>>(
        Wv, Wqkv_t + (size_t)3072 * 2048, 2048, 1024);
    transpose_cvt_kernel<<<dim3(2048 / 16, 2048 / 16), blk, 0, stream>>>(
        Wo, Wo_t, 2048, 2048);
  }
  // 3) fused QKV projection: [4096,2048] @ [2048,4096] -> QKV f32
  gemm_bf16_wmma<<<dim3(4096 / 128, 4096 / 128), 256, 0, stream>>>(
      x_bf, Wqkv_t, QKV, 4096, 4096, 2048);
  // 4) RMSNorm + RoPE + head-major relayout (Q,K) and V relayout
  normrope_kernel<<<dim3(8, B_ * T_), 128, 0, stream>>>(QKV, qw, kw, Qh, Kh, Vh);
  // 5) causal GQA flash attention
  flash_attn_kernel<<<dim3(T_ / 64, B_ * NH_), 128, 0, stream>>>(Qh, Kh, Vh, ctxb);
  // 6) output projection: [4096,2048] @ [2048,2048] -> d_out f32
  gemm_bf16_wmma<<<dim3(2048 / 128, 4096 / 128), 256, 0, stream>>>(
      ctxb, Wo_t, out, 4096, 2048, 2048);
}
